// DINOv3Mask2Former_71305047048431
// MI455X (gfx1250) — matrix operand order repaired and checked by reading, not compiled, not run on hardware
//
#include <hip/hip_runtime.h>
#include <hip/hip_bf16.h>

// ---------------------------------------------------------------------------
// DINO / Mask2Former decoder on MI455X (gfx1250), wave32 + WMMA bf16 path.
// ---------------------------------------------------------------------------

#define LL      6
#define DIM     256
#define NHEAD   8
#define DH      32
#define DFFN    2048
#define NPT     4
#define NLV     3
#define BSZ     8
#define NQQ     300
#define STOT    12096   // 96*96 + 48*48 + 24*24

typedef __attribute__((ext_vector_type(16))) __bf16 v16bf;
typedef __attribute__((ext_vector_type(8)))  __bf16 v8bf;
typedef __attribute__((ext_vector_type(8)))  float  v8f;

// ---------------- wave32 reductions ----------------
__device__ __forceinline__ float wred_sum(float v) {
#pragma unroll
  for (int o = 16; o > 0; o >>= 1) v += __shfl_xor(v, o, 32);
  return v;
}
__device__ __forceinline__ float wred_max(float v) {
#pragma unroll
  for (int o = 16; o > 0; o >>= 1) v = fmaxf(v, __shfl_xor(v, o, 32));
  return v;
}

// convert 8 consecutive f32 (two float4) to packed bf16x8
__device__ __forceinline__ v8bf cvt8_bf16(const float4 a, const float4 b) {
  v8bf o;
  o[0] = (__bf16)a.x; o[1] = (__bf16)a.y; o[2] = (__bf16)a.z; o[3] = (__bf16)a.w;
  o[4] = (__bf16)b.x; o[5] = (__bf16)b.y; o[6] = (__bf16)b.z; o[7] = (__bf16)b.w;
  return o;
}

// ---------------------------------------------------------------------------
// Generic batched GEMM:  C[b] = act(A[b] @ W^T + bias) (+ Res[b])
//   A: [batch, M, K] f32, W: [N, K] f32 shared, C/Res: [batch, M, N] f32.
// flags bit0 = ReLU.  K must be a multiple of 64.
// Block = 256 threads (8 waves). Block tile 64(M) x 128(N); wave tile 16x64.
// K staged 64 at a time (two 32-wide WMMA sub-steps -> 8 WMMAs per barrier
// pair). bf16 staged in LDS via global_load_b128 -> v_cvt_pk_bf16_f32 ->
// ds_store_b128; next K-slab prefetched to L2 (global_prefetch_b8).
// ---------------------------------------------------------------------------
__global__ void __launch_bounds__(256)
gemm_bf16_wmma(const float* __restrict__ A, const float* __restrict__ W,
               const float* __restrict__ bias, const float* __restrict__ Res,
               float* __restrict__ C, int M, int N, int K,
               long sA, long sC, int flags) {
  __shared__ __bf16 sAt[64][72];    // 64x64 tile, padded row stride (144 B)
  __shared__ __bf16 sBt[128][72];   // 128x64 tile

  const int tid  = threadIdx.x;
  const int wave = tid >> 5;
  const int lane = tid & 31;
  const int b    = blockIdx.z;

  const float* Ab = A + (long)b * sA;
  float*       Cb = C + (long)b * sC;
  const float* Rb = Res ? (Res + (long)b * sC) : nullptr;

  const int blockM = blockIdx.y * 64;
  const int blockN = blockIdx.x * 128;
  const int waveM  = (wave >> 1) * 16;   // 0,16,32,48
  const int waveN  = (wave & 1) * 64;    // 0,64

  // staging coordinates: thread owns 16 consecutive K elems of one row
  const int srow = tid >> 2;             // 0..63
  const int scol = (tid & 3) * 16;       // 0,16,32,48

  v8f acc0 = {}, acc1 = {}, acc2 = {}, acc3 = {};

  for (int kk = 0; kk < K; kk += 64) {
    // ---- stage A tile (64x64): 16 f32 -> 16 bf16 per thread ----
    {
      int gm = blockM + srow;
      v8bf p0 = {}, p1 = {};
      if (gm < M) {
        const float4* src = (const float4*)(Ab + (long)gm * K + kk + scol);
        p0 = cvt8_bf16(src[0], src[1]);
        p1 = cvt8_bf16(src[2], src[3]);
      }
      *(v8bf*)&sAt[srow][scol]     = p0;
      *(v8bf*)&sAt[srow][scol + 8] = p1;
    }
    // ---- stage W tile (128x64): two row-slabs per thread ----
#pragma unroll
    for (int i = 0; i < 2; ++i) {
      int r  = srow + i * 64;
      int gn = blockN + r;
      v8bf p0 = {}, p1 = {};
      if (gn < N) {
        const float4* src = (const float4*)(W + (long)gn * K + kk + scol);
        p0 = cvt8_bf16(src[0], src[1]);
        p1 = cvt8_bf16(src[2], src[3]);
      }
      *(v8bf*)&sBt[r][scol]     = p0;
      *(v8bf*)&sBt[r][scol + 8] = p1;
    }
    // prefetch next K-slab into L2 (global_prefetch_b8)
    if (kk + 64 < K) {
      if (blockM + srow < M)
        __builtin_prefetch(Ab + (long)(blockM + srow) * K + kk + 64 + scol, 0, 1);
      if (blockN + srow < N)
        __builtin_prefetch(W + (long)(blockN + srow) * K + kk + 64 + scol, 0, 1);
    }
    __syncthreads();

    // ---- two 32-wide WMMA sub-steps per staged slab ----
#pragma unroll
    for (int ks = 0; ks < 64; ks += 32) {
      v16bf af;
      const int mrow = waveM + (lane & 15);
      const int kA   = ks + ((lane < 16) ? 0 : 16);  // A: lanes<16 -> K0..15
#pragma unroll
      for (int j = 0; j < 8; ++j) {
        af[j]     = sAt[mrow][kA + j];
        af[8 + j] = sAt[mrow][kA + 8 + j];
      }
      const int kB = ks + ((lane < 16) ? 0 : 16);    // B: lanes<16 -> K0..15
      v16bf bf0, bf1, bf2, bf3;
      {
        const int nb = waveN + (lane & 15);
#pragma unroll
        for (int j = 0; j < 16; ++j) {
          bf0[j] = sBt[nb +  0][kB + j];
          bf1[j] = sBt[nb + 16][kB + j];
          bf2[j] = sBt[nb + 32][kB + j];
          bf3[j] = sBt[nb + 48][kB + j];
        }
      }
      acc0 = __builtin_amdgcn_wmma_f32_16x16x32_bf16(false, af, false, bf0, (short)0, acc0, false, false);
      acc1 = __builtin_amdgcn_wmma_f32_16x16x32_bf16(false, af, false, bf1, (short)0, acc1, false, false);
      acc2 = __builtin_amdgcn_wmma_f32_16x16x32_bf16(false, af, false, bf2, (short)0, acc2, false, false);
      acc3 = __builtin_amdgcn_wmma_f32_16x16x32_bf16(false, af, false, bf3, (short)0, acc3, false, false);
    }
    __syncthreads();
  }

  // ---- epilogue: C layout VGPR i -> row i (lanes<16) / i+8 (lanes>=16) ----
  const int mbase = blockM + waveM + ((lane < 16) ? 0 : 8);
  const int ncb   = blockN + waveN + (lane & 15);
  v8f* accs[4] = {&acc0, &acc1, &acc2, &acc3};
#pragma unroll
  for (int t = 0; t < 4; ++t) {
    const int nc = ncb + t * 16;
    if (nc >= N) continue;
    const float bz = bias[nc];
#pragma unroll
    for (int i = 0; i < 8; ++i) {
      int m = mbase + i;
      if (m >= M) continue;
      float v = (*accs[t])[i] + bz;
      if (flags & 1) v = fmaxf(v, 0.0f);
      if (Rb) v += Rb[(long)m * N + nc];
      Cb[(long)m * N + nc] = v;
    }
  }
}

// ---------------------------------------------------------------------------
// LayerNorm over D=256, one wave per row (8 rows / 256-thread block).
// Optionally also emits y2 = LN(x) + addvec (for q = t2 + query_pos).
// ---------------------------------------------------------------------------
__global__ void __launch_bounds__(256)
layernorm_k(const float* __restrict__ x, const float* __restrict__ g,
            const float* __restrict__ bta, float* __restrict__ y,
            const float* __restrict__ addvec, float* __restrict__ y2, int rows) {
  const int row  = blockIdx.x * 8 + (threadIdx.x >> 5);
  const int lane = threadIdx.x & 31;
  if (row >= rows) return;
  const float* xr = x + (long)row * DIM;
  float s = 0.f, sq = 0.f;
  float v[8];
#pragma unroll
  for (int j = 0; j < 8; ++j) {
    v[j] = xr[lane + 32 * j];
    s += v[j]; sq += v[j] * v[j];
  }
  s  = wred_sum(s);
  sq = wred_sum(sq);
  const float mean = s * (1.0f / DIM);
  const float var  = sq * (1.0f / DIM) - mean * mean;
  const float rstd = rsqrtf(var + 1e-5f);
#pragma unroll
  for (int j = 0; j < 8; ++j) {
    int c = lane + 32 * j;
    float o = (v[j] - mean) * rstd * g[c] + bta[c];
    y[(long)row * DIM + c] = o;
    if (y2) y2[(long)row * DIM + c] = o + addvec[(long)row * DIM + c];
  }
}

// ---------------------------------------------------------------------------
// Self-attention: one wave per (b, h, q).  QK: [B][NQ][512], V: [B][NQ][256]
// ---------------------------------------------------------------------------
__global__ void __launch_bounds__(256)
attn_k(const float* __restrict__ QK, const float* __restrict__ V,
       float* __restrict__ O) {
  const int gid  = blockIdx.x * 8 + (threadIdx.x >> 5);
  const int lane = threadIdx.x & 31;
  if (gid >= BSZ * NHEAD * NQQ) return;
  const int b = gid / (NHEAD * NQQ);
  const int h = (gid / NQQ) % NHEAD;
  const int q = gid % NQQ;

  const float scale = 0.17677669529663687f;  // 1/sqrt(32)
  float qreg[DH];
  const float* qp = QK + ((long)(b * NQQ + q) * 512) + h * DH;
#pragma unroll
  for (int d = 0; d < DH; ++d) qreg[d] = qp[d] * scale;

  float lg[10];
#pragma unroll
  for (int j = 0; j < 10; ++j) {
    int k = j * 32 + lane;
    if (k < NQQ) {
      const float* kp = QK + ((long)(b * NQQ + k) * 512) + 256 + h * DH;
      float s = 0.f;
#pragma unroll
      for (int d = 0; d < DH; ++d) s += qreg[d] * kp[d];
      lg[j] = s;
    } else lg[j] = -1e30f;
  }
  float mx = lg[0];
#pragma unroll
  for (int j = 1; j < 10; ++j) mx = fmaxf(mx, lg[j]);
  mx = wred_max(mx);

  float e[10]; float ssum = 0.f;
#pragma unroll
  for (int j = 0; j < 10; ++j) {
    int k = j * 32 + lane;
    e[j] = (k < NQQ) ? __expf(lg[j] - mx) : 0.f;
    ssum += e[j];
  }
  ssum = wred_sum(ssum);
  const float inv = 1.0f / ssum;

  float acc = 0.f;
#pragma unroll 1
  for (int j = 0; j < 10; ++j) {
    for (int sl = 0; sl < 32; ++sl) {
      int k = j * 32 + sl;
      float p = __shfl(e[j], sl, 32);   // uniform trip count across wave
      if (k >= NQQ) break;
      acc += p * V[((long)(b * NQQ + k)) * DIM + h * DH + lane];
    }
  }
  O[((long)(b * NQQ + q)) * DIM + h * DH + lane] = acc * inv;
}

// ---------------------------------------------------------------------------
// Zero masked value rows (key_padding_mask, numpy bool = 1 byte)
// ---------------------------------------------------------------------------
__global__ void mask_k(const unsigned char* __restrict__ mask, float* __restrict__ v) {
  long i = (long)blockIdx.x * 256 + threadIdx.x;
  if (i >= (long)BSZ * STOT) return;
  if (mask[i]) {
    float* p = v + i * DIM;
    for (int d = 0; d < DIM; ++d) p[d] = 0.0f;
  }
}

// ---------------------------------------------------------------------------
// Softmax of attention weights over NLVL*NP = 12, thread per (b, q, h)
// ---------------------------------------------------------------------------
__global__ void awsoftmax_k(float* __restrict__ aw) {
  int idx = blockIdx.x * 256 + threadIdx.x;
  if (idx >= BSZ * NQQ * NHEAD) return;
  int row = idx / NHEAD, h = idx % NHEAD;
  float* p = aw + (long)row * (NHEAD * NLV * NPT) + h * (NLV * NPT);
  float mx = p[0];
#pragma unroll
  for (int t = 1; t < 12; ++t) mx = fmaxf(mx, p[t]);
  float s = 0.f; float e[12];
#pragma unroll
  for (int t = 0; t < 12; ++t) { e[t] = __expf(p[t] - mx); s += e[t]; }
  float inv = 1.0f / s;
#pragma unroll
  for (int t = 0; t < 12; ++t) p[t] = e[t] * inv;
}

// ---------------------------------------------------------------------------
// Deformable sampling: thread per (b, q, h, d). Bilinear gather + weighted sum.
// ---------------------------------------------------------------------------
__global__ void __launch_bounds__(256)
samp_k(const float* __restrict__ vmem, const float* __restrict__ off,
       const float* __restrict__ aw, const float* __restrict__ ref,
       float* __restrict__ outp) {
  long idx = (long)blockIdx.x * 256 + threadIdx.x;
  if (idx >= (long)BSZ * NQQ * DIM) return;
  const int d   = idx & 31;
  const int h   = (idx >> 5) & 7;
  const long bq = idx >> 8;             // b*NQQ + q
  const int b   = (int)(bq / NQQ);

  const int   Hs[3] = {96, 48, 24};
  const int   Ws[3] = {96, 48, 24};
  const int   st[3] = {0, 9216, 11520};

  const float rx = ref[bq * 2 + 0];
  const float ry = ref[bq * 2 + 1];
  float acc = 0.f;

#pragma unroll
  for (int lv = 0; lv < NLV; ++lv) {
    const float Wl = (float)Ws[lv], Hl = (float)Hs[lv];
#pragma unroll
    for (int p = 0; p < NPT; ++p) {
      const long oi = ((bq * NHEAD + h) * NLV + lv) * NPT + p;
      const float ox = off[oi * 2 + 0];
      const float oy = off[oi * 2 + 1];
      const float ix = (rx + ox / Wl) * Wl - 0.5f;
      const float iy = (ry + oy / Hl) * Hl - 0.5f;
      const float w  = aw[(bq * NHEAD + h) * 12 + lv * NPT + p];

      const int x0 = (int)floorf(ix), y0 = (int)floorf(iy);
      const int x1 = x0 + 1,          y1 = y0 + 1;
      const float fx = ix - (float)x0, fy = iy - (float)y0;

      const long base = ((long)b * STOT + st[lv]) * DIM + h * DH + d;
      float s00 = 0.f, s10 = 0.f, s01 = 0.f, s11 = 0.f;
      if (x0 >= 0 && x0 < Ws[lv] && y0 >= 0 && y0 < Hs[lv])
        s00 = vmem[base + (long)(y0 * Ws[lv] + x0) * DIM];
      if (x1 >= 0 && x1 < Ws[lv] && y0 >= 0 && y0 < Hs[lv])
        s10 = vmem[base + (long)(y0 * Ws[lv] + x1) * DIM];
      if (x0 >= 0 && x0 < Ws[lv] && y1 >= 0 && y1 < Hs[lv])
        s01 = vmem[base + (long)(y1 * Ws[lv] + x0) * DIM];
      if (x1 >= 0 && x1 < Ws[lv] && y1 >= 0 && y1 < Hs[lv])
        s11 = vmem[base + (long)(y1 * Ws[lv] + x1) * DIM];

      acc += w * (s00 * (1.f - fx) * (1.f - fy) + s10 * fx * (1.f - fy) +
                  s01 * (1.f - fx) * fy        + s11 * fx * fy);
    }
  }
  outp[bq * DIM + h * DH + d] = acc;
}

// ---------------------------------------------------------------------------
// rp_head tail: ref = sigmoid(h2 @ rp_w2^T + rp_b2), thread per (row, c)
// ---------------------------------------------------------------------------
__global__ void rp2_k(const float* __restrict__ h2, const float* __restrict__ w2,
                      const float* __restrict__ b2, float* __restrict__ refb) {
  int idx = blockIdx.x * 256 + threadIdx.x;
  if (idx >= BSZ * NQQ * 2) return;
  int row = idx >> 1, c = idx & 1;
  const float* hr = h2 + (long)row * DIM;
  const float* wr = w2 + (long)c * DIM;
  float s = 0.f;
#pragma unroll 4
  for (int k = 0; k < DIM; ++k) s += hr[k] * wr[k];
  s += b2[c];
  refb[(long)row * 2 + c] = 1.0f / (1.0f + __expf(-s));
}

// ---------------------------------------------------------------------------
// inter_ref write: broadcast ref over NLVL
// ---------------------------------------------------------------------------
__global__ void interref_k(const float* __restrict__ refb, float* __restrict__ dst) {
  int idx = blockIdx.x * 256 + threadIdx.x;
  if (idx >= BSZ * NQQ * NLV * 2) return;
  int c   = idx & 1;
  int row = idx / (NLV * 2);     // b*NQQ + q
  dst[idx] = refb[(long)row * 2 + c];
}

// ---------------------------------------------------------------------------
extern "C" void kernel_launch(void* const* d_in, const int* in_sizes, int n_in,
                              void* d_out, int out_size, void* d_ws, size_t ws_size,
                              hipStream_t stream) {
  (void)in_sizes; (void)n_in; (void)out_size; (void)ws_size;

  const float* query     = (const float*)d_in[0];
  const float* query_pos = (const float*)d_in[1];
  const float* memory    = (const float*)d_in[2];
  const unsigned char* kmask = (const unsigned char*)d_in[3];
  const float* sa_in_w  = (const float*)d_in[4];
  const float* sa_in_b  = (const float*)d_in[5];
  const float* sa_out_w = (const float*)d_in[6];
  const float* sa_out_b = (const float*)d_in[7];
  const float* n1_g = (const float*)d_in[8];
  const float* n1_b = (const float*)d_in[9];
  const float* n2_g = (const float*)d_in[10];
  const float* n2_b = (const float*)d_in[11];
  const float* n3_g = (const float*)d_in[12];
  const float* n3_b = (const float*)d_in[13];
  const float* vp_w = (const float*)d_in[14];
  const float* vp_b = (const float*)d_in[15];
  const float* so_w = (const float*)d_in[16];
  const float* so_b = (const float*)d_in[17];
  const float* aw_w = (const float*)d_in[18];
  const float* aw_b = (const float*)d_in[19];
  const float* op_w = (const float*)d_in[20];
  const float* op_b = (const float*)d_in[21];
  const float* ff1_w = (const float*)d_in[22];
  const float* ff1_b = (const float*)d_in[23];
  const float* ff2_w = (const float*)d_in[24];
  const float* ff2_b = (const float*)d_in[25];
  const float* fn_g = (const float*)d_in[26];
  const float* fn_b = (const float*)d_in[27];
  const float* rp_w1 = (const float*)d_in[28];
  const float* rp_b1 = (const float*)d_in[29];
  const float* rp_w2 = (const float*)d_in[30];
  const float* rp_b2 = (const float*)d_in[31];

  float* outp = (float*)d_out;
  const long interBase = (long)LL * BSZ * NQQ * DIM;

  // ---- carve workspace ----
  float* ws = (float*)d_ws;
  const long nTok = (long)BSZ * NQQ;
  float* out_s = ws;  ws += nTok * DIM;
  float* t2    = ws;  ws += nTok * DIM;
  float* qkx   = ws;  ws += nTok * DIM;          // t2 + query_pos
  float* qkb   = ws;  ws += nTok * 512;          // q|k projections
  float* vb    = ws;  ws += nTok * DIM;
  float* attno = ws;  ws += nTok * DIM;
  float* vmem  = ws;  ws += (long)BSZ * STOT * DIM;
  float* offb  = ws;  ws += nTok * (NHEAD * NLV * NPT * 2);
  float* awb   = ws;  ws += nTok * (NHEAD * NLV * NPT);
  float* sampo = ws;  ws += nTok * DIM;
  float* ffh   = ws;  ws += nTok * DFFN;
  float* refb  = ws;  ws += nTok * 2;
  float* h2b   = ws;  ws += nTok * DIM;

  auto gemm = [&](const float* A, const float* W, const float* bias,
                  const float* Res, float* C, int M, int N, int K,
                  long sA, long sC, int flags) {
    dim3 g((N + 127) / 128, (M + 63) / 64, BSZ);
    gemm_bf16_wmma<<<g, dim3(256), 0, stream>>>(A, W, bias, Res, C, M, N, K, sA, sC, flags);
  };
  auto lnorm = [&](const float* x, const float* g, const float* b, float* y,
                   const float* addv, float* y2) {
    layernorm_k<<<dim3((int)(nTok / 8)), dim3(256), 0, stream>>>(x, g, b, y, addv, y2, (int)nTok);
  };

  // out = query
  hipMemcpyAsync(out_s, query, nTok * DIM * sizeof(float), hipMemcpyDeviceToDevice, stream);

  // ref = rp_head(query_pos)
  gemm(query_pos, rp_w1, rp_b1, nullptr, h2b, NQQ, DIM, DIM,
       (long)NQQ * DIM, (long)NQQ * DIM, /*relu*/1);
  rp2_k<<<dim3((BSZ * NQQ * 2 + 255) / 256), dim3(256), 0, stream>>>(h2b, rp_w2, rp_b2, refb);

  for (int li = 0; li < LL; ++li) {
    const float* saw  = sa_in_w  + (long)li * 3 * DIM * DIM;
    const float* sab  = sa_in_b  + (long)li * 3 * DIM;
    const float* sow  = sa_out_w + (long)li * DIM * DIM;
    const float* sob  = sa_out_b + (long)li * DIM;
    const float* vpw  = vp_w + (long)li * DIM * DIM;
    const float* vpb  = vp_b + (long)li * DIM;
    const float* sowl = so_w + (long)li * (NHEAD * NLV * NPT * 2) * DIM;
    const float* sobl = so_b + (long)li * (NHEAD * NLV * NPT * 2);
    const float* awwl = aw_w + (long)li * (NHEAD * NLV * NPT) * DIM;
    const float* awbl = aw_b + (long)li * (NHEAD * NLV * NPT);
    const float* opw  = op_w + (long)li * DIM * DIM;
    const float* opb  = op_b + (long)li * DIM;
    const float* f1w  = ff1_w + (long)li * DFFN * DIM;
    const float* f1b  = ff1_b + (long)li * DFFN;
    const float* f2w  = ff2_w + (long)li * DIM * DFFN;
    const float* f2b  = ff2_b + (long)li * DIM;

    // --- self attention block ---
    lnorm(out_s, n1_g + li * DIM, n1_b + li * DIM, t2, query_pos, qkx);
    gemm(qkx, saw, sab, nullptr, qkb, NQQ, 512, DIM, (long)NQQ * DIM, (long)NQQ * 512, 0);
    gemm(t2, saw + 512 * DIM, sab + 512, nullptr, vb, NQQ, DIM, DIM,
         (long)NQQ * DIM, (long)NQQ * DIM, 0);
    attn_k<<<dim3(BSZ * NHEAD * NQQ / 8), dim3(256), 0, stream>>>(qkb, vb, attno);
    gemm(attno, sow, sob, out_s, out_s, NQQ, DIM, DIM, (long)NQQ * DIM, (long)NQQ * DIM, 0);

    // --- deformable attention block ---
    lnorm(out_s, n2_g + li * DIM, n2_b + li * DIM, t2, nullptr, nullptr);
    gemm(memory, vpw, vpb, nullptr, vmem, STOT, DIM, DIM,
         (long)STOT * DIM, (long)STOT * DIM, 0);
    mask_k<<<dim3((int)(((long)BSZ * STOT + 255) / 256)), dim3(256), 0, stream>>>(kmask, vmem);
    gemm(t2, sowl, sobl, nullptr, offb, NQQ, NHEAD * NLV * NPT * 2, DIM,
         (long)NQQ * DIM, (long)NQQ * (NHEAD * NLV * NPT * 2), 0);
    gemm(t2, awwl, awbl, nullptr, awb, NQQ, NHEAD * NLV * NPT, DIM,
         (long)NQQ * DIM, (long)NQQ * (NHEAD * NLV * NPT), 0);
    awsoftmax_k<<<dim3((BSZ * NQQ * NHEAD + 255) / 256), dim3(256), 0, stream>>>(awb);
    samp_k<<<dim3((int)(nTok * DIM / 256)), dim3(256), 0, stream>>>(vmem, offb, awb, refb, sampo);
    gemm(sampo, opw, opb, out_s, out_s, NQQ, DIM, DIM, (long)NQQ * DIM, (long)NQQ * DIM, 0);

    // --- FFN block ---
    lnorm(out_s, n3_g + li * DIM, n3_b + li * DIM, t2, nullptr, nullptr);
    gemm(t2, f1w, f1b, nullptr, ffh, NQQ, DFFN, DIM, (long)NQQ * DIM, (long)NQQ * DFFN, 1);
    gemm(ffh, f2w, f2b, out_s, out_s, NQQ, DIM, DFFN, (long)NQQ * DFFN, (long)NQQ * DIM, 0);

    // --- ref update + outputs ---
    gemm(out_s, rp_w1, rp_b1, nullptr, h2b, NQQ, DIM, DIM,
         (long)NQQ * DIM, (long)NQQ * DIM, 1);
    rp2_k<<<dim3((BSZ * NQQ * 2 + 255) / 256), dim3(256), 0, stream>>>(h2b, rp_w2, rp_b2, refb);

    lnorm(out_s, fn_g, fn_b, outp + (long)li * nTok * DIM, nullptr, nullptr);
    interref_k<<<dim3((BSZ * NQQ * NLV * 2 + 255) / 256), dim3(256), 0, stream>>>(
        refb, outp + interBase + (long)li * nTok * NLV * 2);
  }
}
